// SelfAttention_49546742726773
// MI455X (gfx1250) — matrix operand order repaired and checked by reading, not compiled, not run on hardware
//
#include <hip/hip_runtime.h>
#include <hip/hip_bf16.h>

// ---------------------------------------------------------------------------
// MI455X (gfx1250) self-attention block, bf16 WMMA path.
// cvt(x,Wa,Wp) -> QKV GEMM(+bias,scatter) -> RoPE -> flash attention
// -> projection GEMM(+bias) -> fp32 out.
// All matmuls: v_wmma_f32_16x16x32_bf16, fp32 accumulate.
// GEMM tiling: block = 8 waves, 256(M) x 64(N), K-step 32; each wave owns a
// 32x64 output tile (2 A-frags, 4 register-resident B-frags, 8 WMMA/K-step).
// ---------------------------------------------------------------------------

typedef __bf16 v16bf __attribute__((ext_vector_type(16)));
typedef __bf16 v8bf  __attribute__((ext_vector_type(8)));
typedef float  v8f   __attribute__((ext_vector_type(8)));

#define NHEADS 16
#define HS     128
#define TSEQ   2048
#define CDIM   2048
#define BATCH  2
#define MROWS  (BATCH * TSEQ)     // 4096
#define NQKV   (3 * CDIM)         // 6144

// ---- bf16 helpers ---------------------------------------------------------
__device__ __forceinline__ __bf16 f2bf(float f) {
    union { float f; unsigned u; } in;  in.f = f;
    unsigned u = in.u + (0x7fffu + ((in.u >> 16) & 1u));
    union { unsigned short s; __bf16 b; } out;
    out.s = (unsigned short)(u >> 16);
    return out.b;
}
__device__ __forceinline__ float bf2f(__bf16 b) {
    union { __bf16 b; unsigned short s; } in; in.b = b;
    union { unsigned u; float f; } out;
    out.u = ((unsigned)in.s) << 16;
    return out.f;
}
__device__ __forceinline__ v8bf ld8(const __bf16* p) { return *(const v8bf*)p; }
__device__ __forceinline__ v16bf mk16(v8bf lo, v8bf hi) {
    v16bf r;
#pragma unroll
    for (int j = 0; j < 8; ++j) { r[j] = lo[j]; r[8 + j] = hi[j]; }
    return r;
}
__device__ __forceinline__ v8f zero8() {
    v8f r;
#pragma unroll
    for (int j = 0; j < 8; ++j) r[j] = 0.0f;
    return r;
}
__device__ __forceinline__ v8f wmma_bf16(v16bf a, v16bf b, v8f c) {
    return __builtin_amdgcn_wmma_f32_16x16x32_bf16(false, a, false, b,
                                                   (short)0, c, false, false);
}

// ---------------------------------------------------------------------------
// Kernel 1: fp32 -> bf16 conversion (grid-stride)
// ---------------------------------------------------------------------------
__global__ void cvt_f32_bf16(const float* __restrict__ in,
                             __bf16* __restrict__ out, int n) {
    int i = blockIdx.x * blockDim.x + threadIdx.x;
    int stride = gridDim.x * blockDim.x;
    for (; i < n; i += stride) out[i] = f2bf(in[i]);
}

// ---------------------------------------------------------------------------
// Shared GEMM core: computes a 256x64 block tile of A(bf16,[M][Kd]) @ W(bf16,
// [Kd][ldw]) with Kd=CDIM. Results left in acc[2][4] (v8f each).
// Coop B load: each thread pulls one contiguous v8bf from a W row (b128
// global) and scatters it transposed into LDS (ds_store_b16 x8).
// ---------------------------------------------------------------------------
struct GemmFrag {
    v8f acc[2][4];
};

__device__ __forceinline__ void gemm_core_256x64(
    const __bf16* __restrict__ A, const __bf16* __restrict__ W, int ldw,
    int m0, int n0, __bf16* Bs /* [64*40] */, GemmFrag& fr) {

    const int tid  = threadIdx.x;
    const int wave = tid >> 5;
    const int lane = tid & 31;
    const int n16  = lane & 15;
    const int hi8  = (lane >> 4) * 8;
    const int koff = hi8;                 // 0 or 8
    const int kb16 = hi8 * 2;             // 0 or 16

    const int ldk = tid >> 3;             // 0..31  (k row this thread stages)
    const int ldn = (tid & 7) * 8;        // 0..56  (n offset, 8 elems)

#pragma unroll
    for (int p = 0; p < 2; ++p)
#pragma unroll
        for (int nt = 0; nt < 4; ++nt) fr.acc[p][nt] = zero8();

    for (int k0 = 0; k0 < CDIM; k0 += 32) {
        __syncthreads();
        // Stage B tile transposed: Bs[n][k], row stride 40.
        v8bf wrow = ld8(W + (size_t)(k0 + ldk) * ldw + n0 + ldn);
#pragma unroll
        for (int j = 0; j < 8; ++j) Bs[(ldn + j) * 40 + ldk] = wrow[j];
        if (k0 + 32 < CDIM)
            __builtin_prefetch(W + (size_t)(k0 + 32 + ldk) * ldw + n0 + ldn, 0, 0);
        __syncthreads();

        // B fragments (register resident across both M halves).
        v16bf bf[4];
#pragma unroll
        for (int nt = 0; nt < 4; ++nt) {
            const __bf16* bp = &Bs[(nt * 16 + n16) * 40 + kb16];
            bf[nt] = mk16(ld8(bp), ld8(bp + 8));
        }
        // Two A fragments (wave covers 32 rows).
#pragma unroll
        for (int p = 0; p < 2; ++p) {
            const __bf16* arow =
                A + (size_t)(m0 + wave * 32 + p * 16 + n16) * CDIM + k0;
            v16bf a = mk16(ld8(arow + 2 * koff), ld8(arow + 2 * koff + 8));
#pragma unroll
            for (int nt = 0; nt < 4; ++nt)
                fr.acc[p][nt] = wmma_bf16(a, bf[nt], fr.acc[p][nt]);
        }
    }
}

// ---------------------------------------------------------------------------
// Kernel 2: QKV GEMM + bias, scattered to q/k/v [B*nh, T, hs].
// ---------------------------------------------------------------------------
__global__ __launch_bounds__(256) void qkv_gemm(
    const __bf16* __restrict__ xb, const __bf16* __restrict__ Wa,
    const float* __restrict__ bias,
    __bf16* __restrict__ q, __bf16* __restrict__ k, __bf16* __restrict__ v) {

    __shared__ alignas(32) __bf16 Bs[64 * 40];
    GemmFrag fr;
    const int m0 = blockIdx.y * 256;
    const int n0 = blockIdx.x * 64;
    gemm_core_256x64(xb, Wa, NQKV, m0, n0, Bs, fr);

    const int wave = threadIdx.x >> 5;
    const int lane = threadIdx.x & 31;
    const int n16  = lane & 15;
    const int hi8  = (lane >> 4) * 8;

#pragma unroll
    for (int nt = 0; nt < 4; ++nt) {
        int gn = n0 + nt * 16 + n16;
        int which = gn >> 11;             // 0=q 1=k 2=v
        int c = gn & (CDIM - 1);
        int h = c >> 7;
        int d = c & (HS - 1);
        __bf16* dst = (which == 0) ? q : ((which == 1) ? k : v);
        float bval = bias[gn];
#pragma unroll
        for (int p = 0; p < 2; ++p)
#pragma unroll
            for (int r = 0; r < 8; ++r) {
                int gm = m0 + wave * 32 + p * 16 + hi8 + r;
                int bb = gm >> 11;
                int t  = gm & (TSEQ - 1);
                dst[(((size_t)(bb * NHEADS + h)) * TSEQ + t) * HS + d] =
                    f2bf(fr.acc[p][nt][r] + bval);
            }
    }
}

// ---------------------------------------------------------------------------
// Kernel 3: RoPE, in place on q and k.
// ---------------------------------------------------------------------------
__global__ void rope_kernel(__bf16* __restrict__ q, __bf16* __restrict__ k) {
    const int per_buf = BATCH * NHEADS * TSEQ * (HS / 2);
    int idx = blockIdx.x * blockDim.x + threadIdx.x;
    int stride = gridDim.x * blockDim.x;
    for (; idx < 2 * per_buf; idx += stride) {
        int buf = idx / per_buf;
        int rem = idx - buf * per_buf;
        int bh = rem / (TSEQ * (HS / 2));
        int r2 = rem - bh * (TSEQ * (HS / 2));
        int t = r2 >> 6;
        int d = r2 & 63;
        __bf16* base = (buf ? k : q) + ((size_t)bh * TSEQ + t) * HS;
        float inv = __powf(10000.0f, -(float)d * (1.0f / 64.0f));
        float ang = (float)t * inv;
        float s, c;
        __sincosf(ang, &s, &c);
        float x1 = bf2f(base[d]);
        float x2 = bf2f(base[d + 64]);
        base[d]      = f2bf(x1 * c - x2 * s);
        base[d + 64] = f2bf(x2 * c + x1 * s);
    }
}

// ---------------------------------------------------------------------------
// Kernel 4: causal flash attention.
// Grid: (B*nh, T/64). Block: 128 threads = 4 waves, each wave = 16 q rows.
// ---------------------------------------------------------------------------
__global__ __launch_bounds__(128) void flash_attn(
    const __bf16* __restrict__ q, const __bf16* __restrict__ k,
    const __bf16* __restrict__ v, __bf16* __restrict__ yatt) {

    __shared__ alignas(32) __bf16 Ks[32 * 128];      // [key][d]
    __shared__ alignas(32) __bf16 Vt[128 * 32];      // [d][key]
    __shared__ alignas(32) __bf16 Pw[4][16 * 32];    // per-wave P staging

    const int tid  = threadIdx.x;
    const int wave = tid >> 5;
    const int lane = tid & 31;
    const int n16  = lane & 15;
    const int hi8  = (lane >> 4) * 8;
    const int koff = hi8;
    const int kb16 = hi8 * 2;

    const int bh = blockIdx.x;
    const size_t hoff = (size_t)bh * TSEQ * HS;
    const int qrow0 = blockIdx.y * 64 + wave * 16;

    // Coop-load decomposition: 512 v8bf chunks over 128 threads = 4 each.
    // chunk c: key row kc = c>>4, d8 = (c&15)*8.
    const __bf16* qrow = q + hoff + (size_t)(qrow0 + n16) * HS;
    v16bf qf[4];
#pragma unroll
    for (int kk = 0; kk < 4; ++kk)
        qf[kk] = mk16(ld8(qrow + kk * 32 + 2 * koff), ld8(qrow + kk * 32 + 2 * koff + 8));

    float mrow[8], lrow[8];
    v8f o[8];
#pragma unroll
    for (int r = 0; r < 8; ++r) { mrow[r] = -3.0e38f; lrow[r] = 0.0f; }
#pragma unroll
    for (int i = 0; i < 8; ++i) o[i] = zero8();

    const float scale = 0.08838834764831845f;  // 1/sqrt(128)
    const int nchunk = (blockIdx.y + 1) * 2;   // uniform causal bound

    for (int ch = 0; ch < nchunk; ++ch) {
        const int kt0 = ch * 32;
        __syncthreads();
#pragma unroll
        for (int i = 0; i < 4; ++i) {
            int c  = tid + i * 128;
            int kc = c >> 4;
            int d8 = (c & 15) * 8;
            const size_t gsrc = hoff + (size_t)(kt0 + kc) * HS + d8;
            // K: row-major b128 copy.
            *(v8bf*)&Ks[kc * 128 + d8] = ld8(k + gsrc);
            // V: contiguous global load, transposed LDS scatter.
            v8bf vv = ld8(v + gsrc);
#pragma unroll
            for (int j = 0; j < 8; ++j) Vt[(d8 + j) * 32 + kc] = vv[j];
        }
        __syncthreads();

        // S = (Q K^T) * scale, two 16-key tiles.
        v8f sa[2];
#pragma unroll
        for (int nt = 0; nt < 2; ++nt) {
            v8f accs = zero8();
#pragma unroll
            for (int kk = 0; kk < 4; ++kk) {
                const __bf16* bp = &Ks[(nt * 16 + n16) * 128 + kk * 32 + kb16];
                v16bf bfrag = mk16(ld8(bp), ld8(bp + 8));
                accs = wmma_bf16(qf[kk], bfrag, accs);
            }
            sa[nt] = accs;
        }

        // Mask + online softmax update.
#pragma unroll
        for (int r = 0; r < 8; ++r) {
            int row = qrow0 + hi8 + r;
            float s0 = sa[0][r] * scale;
            float s1 = sa[1][r] * scale;
            if (kt0 + n16 > row)      s0 = -3.0e38f;
            if (kt0 + 16 + n16 > row) s1 = -3.0e38f;
            float pm = fmaxf(s0, s1);
            pm = fmaxf(pm, __shfl_xor(pm, 1, 32));
            pm = fmaxf(pm, __shfl_xor(pm, 2, 32));
            pm = fmaxf(pm, __shfl_xor(pm, 4, 32));
            pm = fmaxf(pm, __shfl_xor(pm, 8, 32));
            float mnew = fmaxf(mrow[r], pm);
            float alpha = __expf(mrow[r] - mnew);
            float p0 = __expf(s0 - mnew);
            float p1 = __expf(s1 - mnew);
            float ps = p0 + p1;
            ps += __shfl_xor(ps, 1, 32);
            ps += __shfl_xor(ps, 2, 32);
            ps += __shfl_xor(ps, 4, 32);
            ps += __shfl_xor(ps, 8, 32);
            lrow[r] = lrow[r] * alpha + ps;
            mrow[r] = mnew;
#pragma unroll
            for (int nt2 = 0; nt2 < 8; ++nt2) o[nt2][r] *= alpha;
            sa[0][r] = p0;
            sa[1][r] = p1;
        }

        // Re-layout P (C-frag -> A-frag) through per-wave LDS.
        __bf16* pw = &Pw[wave][0];
#pragma unroll
        for (int nt = 0; nt < 2; ++nt)
#pragma unroll
            for (int r = 0; r < 8; ++r)
                pw[(hi8 + r) * 32 + nt * 16 + n16] = f2bf(sa[nt][r]);
        asm volatile("s_wait_dscnt 0x0" ::: "memory");  // same-wave LDS RAW
        v16bf pA = mk16(ld8(pw + n16 * 32 + 2 * koff), ld8(pw + n16 * 32 + 2 * koff + 8));

        // O += P @ V over the 8 hs tiles.
#pragma unroll
        for (int nt2 = 0; nt2 < 8; ++nt2) {
            const __bf16* vp = &Vt[(nt2 * 16 + n16) * 32 + kb16];
            v16bf vb = mk16(ld8(vp), ld8(vp + 8));
            o[nt2] = wmma_bf16(pA, vb, o[nt2]);
        }
    }

    // Normalize and write to [B, T, C] bf16 for the projection GEMM.
    const int bb = bh >> 4;
    const int h  = bh & (NHEADS - 1);
    float invl[8];
#pragma unroll
    for (int r = 0; r < 8; ++r) invl[r] = 1.0f / lrow[r];
#pragma unroll
    for (int nt2 = 0; nt2 < 8; ++nt2)
#pragma unroll
        for (int r = 0; r < 8; ++r) {
            int t = qrow0 + hi8 + r;
            yatt[((size_t)(bb * TSEQ + t)) * CDIM + h * HS + nt2 * 16 + n16] =
                f2bf(o[nt2][r] * invl[r]);
        }
}

// ---------------------------------------------------------------------------
// Kernel 5: projection GEMM. out = yatt @ Wp + b_proj (fp32 output).
// ---------------------------------------------------------------------------
__global__ __launch_bounds__(256) void proj_gemm(
    const __bf16* __restrict__ yatt, const __bf16* __restrict__ Wp,
    const float* __restrict__ bias, float* __restrict__ out) {

    __shared__ alignas(32) __bf16 Bs[64 * 40];
    GemmFrag fr;
    const int m0 = blockIdx.y * 256;
    const int n0 = blockIdx.x * 64;
    gemm_core_256x64(yatt, Wp, CDIM, m0, n0, Bs, fr);

    const int wave = threadIdx.x >> 5;
    const int lane = threadIdx.x & 31;
    const int n16  = lane & 15;
    const int hi8  = (lane >> 4) * 8;

#pragma unroll
    for (int nt = 0; nt < 4; ++nt) {
        int gn = n0 + nt * 16 + n16;
        float bval = bias[gn];
#pragma unroll
        for (int p = 0; p < 2; ++p)
#pragma unroll
            for (int r = 0; r < 8; ++r) {
                int gm = m0 + wave * 32 + p * 16 + hi8 + r;
                out[(size_t)gm * CDIM + gn] = fr.acc[p][nt][r] + bval;
            }
    }
}

// ---------------------------------------------------------------------------
// Launch
// ---------------------------------------------------------------------------
extern "C" void kernel_launch(void* const* d_in, const int* in_sizes, int n_in,
                              void* d_out, int out_size, void* d_ws, size_t ws_size,
                              hipStream_t stream) {
    const float* x      = (const float*)d_in[0];
    const float* W_attn = (const float*)d_in[1];
    const float* b_attn = (const float*)d_in[2];
    const float* W_proj = (const float*)d_in[3];
    const float* b_proj = (const float*)d_in[4];
    float* out = (float*)d_out;

    // Workspace layout (bf16 buffers), ~112 MB total.
    char* ws = (char*)d_ws;
    size_t off = 0;
    __bf16* xb = (__bf16*)(ws + off); off += (size_t)MROWS * CDIM * 2;
    __bf16* Wa = (__bf16*)(ws + off); off += (size_t)CDIM * NQKV * 2;
    __bf16* Wp = (__bf16*)(ws + off); off += (size_t)CDIM * CDIM * 2;
    __bf16* qb = (__bf16*)(ws + off); off += (size_t)BATCH * NHEADS * TSEQ * HS * 2;
    __bf16* kb = (__bf16*)(ws + off); off += (size_t)BATCH * NHEADS * TSEQ * HS * 2;
    __bf16* vb = (__bf16*)(ws + off); off += (size_t)BATCH * NHEADS * TSEQ * HS * 2;
    __bf16* ya = (__bf16*)(ws + off); off += (size_t)MROWS * CDIM * 2;

    cvt_f32_bf16<<<dim3(4096), dim3(256), 0, stream>>>(x, xb, MROWS * CDIM);
    cvt_f32_bf16<<<dim3(4096), dim3(256), 0, stream>>>(W_attn, Wa, CDIM * NQKV);
    cvt_f32_bf16<<<dim3(4096), dim3(256), 0, stream>>>(W_proj, Wp, CDIM * CDIM);

    qkv_gemm<<<dim3(NQKV / 64, MROWS / 256), dim3(256), 0, stream>>>(
        xb, Wa, b_attn, qb, kb, vb);

    rope_kernel<<<dim3(8192), dim3(256), 0, stream>>>(qb, kb);

    flash_attn<<<dim3(BATCH * NHEADS, TSEQ / 64), dim3(128), 0, stream>>>(
        qb, kb, vb, ya);

    proj_gemm<<<dim3(CDIM / 64, MROWS / 256), dim3(256), 0, stream>>>(
        ya, Wp, b_proj, out);
}
